// QuantizedWeightLayer_45664092291229
// MI455X (gfx1250) — matrix operand order repaired and checked by reading, not compiled, not run on hardware
//
#include <hip/hip_runtime.h>

typedef __attribute__((ext_vector_type(16))) _Float16 v16h;
typedef __attribute__((ext_vector_type(8)))  _Float16 v8h;
typedef __attribute__((ext_vector_type(8)))  float    v8f;

#define M_TOTAL 8192   // B * S = 4 * 2048
#define N_TOTAL 4096   // UNITS
#define K_TOTAL 4096   // D
#define BLK_M   128
#define BLK_N   128

// ---------------------------------------------------------------------------
// Kernel 1: one-shot W dequant-layout kernel: int8 W[K][N] -> f16 Wt[N][K]
// 64x64 tiles through padded LDS; coalesced uint4 reads, coalesced b128 writes.
// ---------------------------------------------------------------------------
__global__ __launch_bounds__(256) void convert_transpose_w(
    const signed char* __restrict__ W, _Float16* __restrict__ Wt)
{
    __shared__ _Float16 t[64][72];        // 72-half rows: 144B, 16B-aligned, bank-spread

    const int k0  = blockIdx.x * 64;
    const int n0  = blockIdx.y * 64;
    const int tid = threadIdx.x;
    const int r   = tid >> 2;             // 0..63
    const int c4  = (tid & 3) << 4;       // 0,16,32,48

    // read W[k0+r][n0+c4 .. +15]: 16 int8, fully coalesced
    const uint4 v = *(const uint4*)(W + (size_t)(k0 + r) * N_TOTAL + n0 + c4);
    const unsigned int wv[4] = { v.x, v.y, v.z, v.w };
#pragma unroll
    for (int j = 0; j < 4; ++j) {
#pragma unroll
        for (int i = 0; i < 4; ++i) {
            const signed char ch = (signed char)(wv[j] >> (8 * i));
            t[c4 + j * 4 + i][r] = (_Float16)(short)ch;   // transposed scatter
        }
    }
    __syncthreads();

    // write Wt[n0+r][k0+c4 .. +15]: two aligned b128 stores, coalesced
    const v8h* src = (const v8h*)(&t[r][c4]);
    v8h* dst = (v8h*)(Wt + (size_t)(n0 + r) * K_TOTAL + k0 + c4);
    dst[0] = src[0];
    dst[1] = src[1];
}

// ---------------------------------------------------------------------------
// Kernel 2 (fast path): GEMM with pre-converted Wt[N][K] f16.
// A staged fp32->f16 through double-buffered LDS (BLK_K=64, 1 barrier / 16 WMMA);
// B fragments loaded straight from global (L2-resident) into DISTINCT registers
// up-front so load latency pipelines across the WMMA stream.
// ---------------------------------------------------------------------------
#define BLK_K2 64
#define ROW_H2 72   // 64 data + 8 pad halves; 144B rows (multiple of 16B)

__global__ __launch_bounds__(256) void qgemm_wmma_f16_pre(
    const float* __restrict__ X,          // [M_TOTAL, K_TOTAL] fp32
    const _Float16* __restrict__ Wt,      // [N_TOTAL, K_TOTAL] f16 (pre-transposed)
    const float* __restrict__ scale_p,
    float* __restrict__ Y)                // [M_TOTAL, N_TOTAL] fp32
{
    __shared__ _Float16 As[2][BLK_M * ROW_H2];   // 2 x 18432 B

    const int tid  = threadIdx.x;
    const int lane = tid & 31;
    const int wave = tid >> 5;
    const int wm   = (wave & 3) * 32;    // 4 waves along M -> 128
    const int wn   = (wave >> 2) * 64;   // 2 waves along N -> 128 (wave tile 32x64)

    const int bm = blockIdx.x * BLK_M;
    const int bn = blockIdx.y * BLK_N;

    const int fr = lane & 15;            // row (A) / column (B,C,D)
    const int fh = lane >> 4;            // 0/1: K-half (A,B) / M-half (C,D)

    const float scale = scale_p[0];

    const v8f zero8 = {0.f, 0.f, 0.f, 0.f, 0.f, 0.f, 0.f, 0.f};
    v8f acc[2][4] = { { zero8, zero8, zero8, zero8 },
                      { zero8, zero8, zero8, zero8 } };

    // A staging: 128 rows x 64 floats per slab; thread: row = tid/2, 32 floats
    const int a_row  = tid >> 1;
    const int a_part = (tid & 1) << 5;   // 0 or 32 floats
    const float* a_src = X + (size_t)(bm + a_row) * K_TOTAL + a_part;

    auto stageA = [&](int buf, int k0) {
        const float4* s4 = (const float4*)(a_src + k0);
#pragma unroll
        for (int q = 0; q < 2; ++q) {
            float4 f0 = s4[q * 4 + 0], f1 = s4[q * 4 + 1];
            float4 f2 = s4[q * 4 + 2], f3 = s4[q * 4 + 3];
            v8h h0, h1;
            h0[0] = (_Float16)f0.x; h0[1] = (_Float16)f0.y;
            h0[2] = (_Float16)f0.z; h0[3] = (_Float16)f0.w;
            h0[4] = (_Float16)f1.x; h0[5] = (_Float16)f1.y;
            h0[6] = (_Float16)f1.z; h0[7] = (_Float16)f1.w;
            h1[0] = (_Float16)f2.x; h1[1] = (_Float16)f2.y;
            h1[2] = (_Float16)f2.z; h1[3] = (_Float16)f2.w;
            h1[4] = (_Float16)f3.x; h1[5] = (_Float16)f3.y;
            h1[6] = (_Float16)f3.z; h1[7] = (_Float16)f3.w;
            v8h* dst = (v8h*)(&As[buf][a_row * ROW_H2 + a_part + q * 16]);
            dst[0] = h0;
            dst[1] = h1;
        }
        if (k0 + BLK_K2 < K_TOTAL)
            __builtin_prefetch(a_src + k0 + BLK_K2, 0, 3);
    };

    // B fragment base: lane L -> column bn+wn+ni*16+(L%16), K = k + (L/16)*8 ..+7, k + 16 + (L/16)*8 ..+7
    const _Float16* b_base = Wt + (size_t)(bn + wn + fr) * K_TOTAL + fh * 8;

    stageA(0, 0);
    __syncthreads();

    for (int k0 = 0; k0 < K_TOTAL; k0 += BLK_K2) {
        const int buf = (k0 >> 6) & 1;

        // Issue ALL 8 B-fragment loads first (distinct regs -> pipelined waits)
        v16h bf[2][4];
#pragma unroll
        for (int kk2 = 0; kk2 < 2; ++kk2) {
#pragma unroll
            for (int ni = 0; ni < 4; ++ni) {
                const _Float16* bp =
                    b_base + (size_t)ni * 16 * K_TOTAL + (k0 + kk2 * 32);
                v8h lo = *(const v8h*)bp;
                v8h hi = *(const v8h*)(bp + 16);
#pragma unroll
                for (int i = 0; i < 8; ++i) {
                    bf[kk2][ni][i]     = lo[i];
                    bf[kk2][ni][8 + i] = hi[i];
                }
            }
        }

        // Stage next A slab into the other buffer while computing this one
        if (k0 + BLK_K2 < K_TOTAL)
            stageA(buf ^ 1, k0 + BLK_K2);

#pragma unroll
        for (int kk2 = 0; kk2 < 2; ++kk2) {
            // A 16x32 f16 fragment: halves 0..7 = K[(L/16)*8+0..7],
            //                        halves 8..15 = K[16+(L/16)*8+0..7]
            v16h af[2];
#pragma unroll
            for (int mi = 0; mi < 2; ++mi) {
                const _Float16* ap =
                    &As[buf][(wm + mi * 16 + fr) * ROW_H2 + kk2 * 32 + fh * 8];
                v8h lo = *(const v8h*)ap;
                v8h hi = *(const v8h*)(ap + 16);
#pragma unroll
                for (int i = 0; i < 8; ++i) { af[mi][i] = lo[i]; af[mi][8 + i] = hi[i]; }
            }
#pragma unroll
            for (int mi = 0; mi < 2; ++mi) {
#pragma unroll
                for (int ni = 0; ni < 4; ++ni) {
                    acc[mi][ni] = __builtin_amdgcn_wmma_f32_16x16x32_f16(
                        false, af[mi], false, bf[kk2][ni], (short)0, acc[mi][ni],
                        false, false);
                }
            }
        }
        __syncthreads();
    }

    // Epilogue: C/D layout: VGPR r -> M = r + 8*(lane/16), N = lane%16.
    // Non-temporal stores: Y is write-once; keep L2 free for Wt residency.
#pragma unroll
    for (int mi = 0; mi < 2; ++mi) {
#pragma unroll
        for (int ni = 0; ni < 4; ++ni) {
            const int col = bn + wn + ni * 16 + fr;
            float* out = Y + (size_t)(bm + wm + mi * 16 + fh * 8) * N_TOTAL + col;
#pragma unroll
            for (int r = 0; r < 8; ++r)
                __builtin_nontemporal_store(scale * acc[mi][ni][r],
                                            out + (size_t)r * N_TOTAL);
        }
    }
}

// ---------------------------------------------------------------------------
// Fallback kernel (no workspace): self-contained version.
// ---------------------------------------------------------------------------
#define BLK_K   32
#define ROW_H   40

__global__ __launch_bounds__(256) void qgemm_wmma_f16(
    const float* __restrict__ X, const signed char* __restrict__ W,
    const float* __restrict__ scale_p, float* __restrict__ Y)
{
    __shared__ _Float16 As[2][BLK_M * ROW_H];
    __shared__ _Float16 Bt[2][BLK_N * ROW_H];

    const int tid  = threadIdx.x;
    const int lane = tid & 31;
    const int wave = tid >> 5;
    const int wm   = (wave & 3) * 32;
    const int wn   = (wave >> 2) * 64;

    const int bm = blockIdx.x * BLK_M;
    const int bn = blockIdx.y * BLK_N;

    const int fr = lane & 15;
    const int fh = lane >> 4;

    const float scale = scale_p[0];

    const v8f zero8 = {0.f, 0.f, 0.f, 0.f, 0.f, 0.f, 0.f, 0.f};
    v8f acc[2][4] = { { zero8, zero8, zero8, zero8 },
                      { zero8, zero8, zero8, zero8 } };

    const int a_row  = tid >> 1;
    const int a_part = (tid & 1) << 4;
    const int b_krow = tid >> 3;
    const int b_col  = (tid & 7) << 4;

    const float* a_src_base = X + (size_t)(bm + a_row) * K_TOTAL + a_part;
    const signed char* b_src_base = W + (size_t)b_krow * N_TOTAL + bn + b_col;

    auto stageA = [&](int buf, int k0) {
        const float4* s4 = (const float4*)(a_src_base + k0);
        float4 f0 = s4[0], f1 = s4[1], f2 = s4[2], f3 = s4[3];
        v8h h0, h1;
        h0[0] = (_Float16)f0.x; h0[1] = (_Float16)f0.y;
        h0[2] = (_Float16)f0.z; h0[3] = (_Float16)f0.w;
        h0[4] = (_Float16)f1.x; h0[5] = (_Float16)f1.y;
        h0[6] = (_Float16)f1.z; h0[7] = (_Float16)f1.w;
        h1[0] = (_Float16)f2.x; h1[1] = (_Float16)f2.y;
        h1[2] = (_Float16)f2.z; h1[3] = (_Float16)f2.w;
        h1[4] = (_Float16)f3.x; h1[5] = (_Float16)f3.y;
        h1[6] = (_Float16)f3.z; h1[7] = (_Float16)f3.w;
        v8h* dst = (v8h*)(&As[buf][a_row * ROW_H + a_part]);
        dst[0] = h0;
        dst[1] = h1;
    };
    auto stageB = [&](int buf, int k0) {
        const uint4 v = *(const uint4*)(b_src_base + (size_t)k0 * N_TOTAL);
        const unsigned int wv[4] = { v.x, v.y, v.z, v.w };
#pragma unroll
        for (int j = 0; j < 4; ++j)
#pragma unroll
            for (int i = 0; i < 4; ++i) {
                const signed char c = (signed char)(wv[j] >> (8 * i));
                Bt[buf][(b_col + j * 4 + i) * ROW_H + b_krow] = (_Float16)(short)c;
            }
    };

    stageA(0, 0);
    stageB(0, 0);
    __syncthreads();

    for (int k0 = 0; k0 < K_TOTAL; k0 += BLK_K) {
        const int buf = (k0 >> 5) & 1;
        if (k0 + BLK_K < K_TOTAL) {
            stageA(buf ^ 1, k0 + BLK_K);
            stageB(buf ^ 1, k0 + BLK_K);
        }

        v16h af[2];
#pragma unroll
        for (int mi = 0; mi < 2; ++mi) {
            const _Float16* ap = &As[buf][(wm + mi * 16 + fr) * ROW_H + fh * 8];
            v8h lo = *(const v8h*)ap;
            v8h hi = *(const v8h*)(ap + 16);
#pragma unroll
            for (int i = 0; i < 8; ++i) { af[mi][i] = lo[i]; af[mi][8 + i] = hi[i]; }
        }
        v16h bfr[4];
#pragma unroll
        for (int ni = 0; ni < 4; ++ni) {
            const _Float16* bp = &Bt[buf][(wn + ni * 16 + fr) * ROW_H + fh * 8];
            v8h lo = *(const v8h*)bp;
            v8h hi = *(const v8h*)(bp + 16);
#pragma unroll
            for (int i = 0; i < 8; ++i) { bfr[ni][i] = lo[i]; bfr[ni][8 + i] = hi[i]; }
        }

#pragma unroll
        for (int mi = 0; mi < 2; ++mi)
#pragma unroll
            for (int ni = 0; ni < 4; ++ni)
                acc[mi][ni] = __builtin_amdgcn_wmma_f32_16x16x32_f16(
                    false, af[mi], false, bfr[ni], (short)0, acc[mi][ni],
                    false, false);

        __syncthreads();
    }

#pragma unroll
    for (int mi = 0; mi < 2; ++mi)
#pragma unroll
        for (int ni = 0; ni < 4; ++ni) {
            const int col = bn + wn + ni * 16 + fr;
            float* out = Y + (size_t)(bm + wm + mi * 16 + fh * 8) * N_TOTAL + col;
#pragma unroll
            for (int r = 0; r < 8; ++r)
                __builtin_nontemporal_store(scale * acc[mi][ni][r],
                                            out + (size_t)r * N_TOTAL);
        }
}

// ---------------------------------------------------------------------------
extern "C" void kernel_launch(void* const* d_in, const int* in_sizes, int n_in,
                              void* d_out, int out_size, void* d_ws, size_t ws_size,
                              hipStream_t stream) {
    const float*       x     = (const float*)d_in[0];
    const signed char* w     = (const signed char*)d_in[1];   // int8 weights
    const float*       scale = (const float*)d_in[2];
    float*             y     = (float*)d_out;

    const size_t wt_bytes = (size_t)N_TOTAL * K_TOTAL * sizeof(_Float16); // 32 MiB

    if (ws_size >= wt_bytes) {
        _Float16* wt = (_Float16*)d_ws;
        convert_transpose_w<<<dim3(K_TOTAL / 64, N_TOTAL / 64), 256, 0, stream>>>(w, wt);
        qgemm_wmma_f16_pre<<<dim3(M_TOTAL / BLK_M, N_TOTAL / BLK_N), 256, 0, stream>>>(
            x, wt, scale, y);
    } else {
        qgemm_wmma_f16<<<dim3(M_TOTAL / BLK_M, N_TOTAL / BLK_N), 256, 0, stream>>>(
            x, w, scale, y);
    }

    (void)in_sizes; (void)n_in; (void)out_size;
}
